// LocalDownSample_68719477129
// MI455X (gfx1250) — matrix operand order repaired and checked by reading, not compiled, not run on hardware
//
#include <hip/hip_runtime.h>
#include <hip/hip_bf16.h>
#include <math.h>

// ---------------- problem constants ----------------
#define BB 2
#define CC 128
#define NN 8192
#define KK 32
#define MM 4096

typedef __attribute__((ext_vector_type(16))) __bf16 v16bf;
typedef __attribute__((ext_vector_type(8)))  __bf16 v8bf;
typedef __attribute__((ext_vector_type(8)))  float  v8f;
typedef __attribute__((ext_vector_type(4)))  float  v4f;

// ---- WMMA helpers (16x16x32 bf16, f32 accum) ----
__device__ __forceinline__ v8f wmma_bf16(v16bf a, v16bf b, v8f c) {
  return __builtin_amdgcn_wmma_f32_16x16x32_bf16(false, a, false, b, (short)0, c,
                                                 false, false);
}

// A fragment: 16x32 tile, lane holds row (lane&15); K chunks [k0..k0+7],[k0+16..k0+23], k0=(lane>>4)*8
__device__ __forceinline__ v16bf load_frag_a(const __bf16* rowptr, int lane) {
  int k0 = (lane >> 4) << 3;
  v8bf lo = *(const v8bf*)(rowptr + k0);
  v8bf hi = *(const v8bf*)(rowptr + k0 + 16);
  v16bf r;
#pragma unroll
  for (int i = 0; i < 8; ++i) { r[i] = lo[i]; r[i + 8] = hi[i]; }
  return r;
}

// B fragment: 32x16 tile; lane holds column (lane&15); K chunks [klo..klo+7],[klo+16..klo+23], klo=(lane>>4)*8.
// colptr = pointer to the (row-major) source row that forms this B column.
__device__ __forceinline__ v16bf load_frag_b(const __bf16* colptr, int lane) {
  int klo = (lane >> 4) << 3;
  v8bf lo = *(const v8bf*)(colptr + klo);
  v8bf hi = *(const v8bf*)(colptr + klo + 16);
  v16bf r;
#pragma unroll
  for (int i = 0; i < 8; ++i) { r[i] = lo[i]; r[i + 8] = hi[i]; }
  return r;
}

// ---------------- kernel A: transpose x -> pcd (f32 + bf16) ----------------
__global__ void prep_kernel(const float* __restrict__ x, float* __restrict__ pcdf,
                            __bf16* __restrict__ pcdh) {
  size_t t = (size_t)blockIdx.x * blockDim.x + threadIdx.x;  // over B*C*N
  size_t b = t / ((size_t)CC * NN);
  size_t rem = t % ((size_t)CC * NN);
  size_t c = rem / NN, n = rem % NN;
  float v = x[t];
  size_t o = ((size_t)b * NN + n) * CC + c;
  pcdf[o] = v;
  pcdh[o] = (__bf16)v;
}

// ---------------- kernel: squared norms ----------------
__global__ void norm_kernel(const float* __restrict__ pcdf, float* __restrict__ sq) {
  int t = blockIdx.x * blockDim.x + threadIdx.x;  // over B*N
  const float* row = pcdf + (size_t)t * CC;
  float s = 0.f;
#pragma unroll 8
  for (int c = 0; c < CC; ++c) { float v = row[c]; s += v * v; }
  sq[t] = s;
}

// ---------------- kernel: Wqk = (Wk^T Wq)/sqrt(C)  and Wv -> bf16 ----------------
__global__ void prep_w_kernel(const float* __restrict__ Wq, const float* __restrict__ Wk,
                              const float* __restrict__ Wv, __bf16* __restrict__ wqk_bf,
                              __bf16* __restrict__ wv_bf) {
  int t = blockIdx.x * blockDim.x + threadIdx.x;  // over 128*128
  int i = t >> 7, j = t & 127;
  float s = 0.f;
#pragma unroll 8
  for (int d = 0; d < CC; ++d) s += Wk[d * CC + i] * Wq[d * CC + j];
  wqk_bf[t] = (__bf16)(s * 0.08838834764831845f);  // 1/sqrt(128)
  wv_bf[t] = (__bf16)Wv[t];
}

// ---------------- kernel B: fused distance GEMM + top-32 KNN ----------------
// block = 512 threads = 16 waves; block owns 16 rows; wave w computes col tile w of each
// 256-col stripe via WMMA, then wave w maintains top-32 for row w.
__global__ void __launch_bounds__(512) knn_kernel(const __bf16* __restrict__ pcdh,
                                                  const float* __restrict__ sq,
                                                  int* __restrict__ knn_idx) {
  __shared__ float dist[16][257];  // padded to dodge bank conflicts
  __shared__ float sqrow[16];

  int b = blockIdx.x / (NN / 16);
  int rowBase = (blockIdx.x % (NN / 16)) * 16;
  int wave = threadIdx.x >> 5;
  int lane = threadIdx.x & 31;

  if (threadIdx.x < 16) sqrow[threadIdx.x] = sq[b * NN + rowBase + threadIdx.x];
  __syncthreads();

  const __bf16* base = pcdh + (size_t)b * NN * CC;
  int arow = rowBase + (lane & 15);
  int rhalf = (lane >> 4) << 3;

  // per-lane top-32 (lane owns candidate columns ≡ lane (mod 32) of row `wave`)
  float tval[KK];
  int tidx[KK];
#pragma unroll
  for (int t = 0; t < KK; ++t) { tval[t] = -3.4e38f; tidx[t] = -1; }
  float minV = -3.4e38f;
  int minP = 0;

  for (int iter = 0; iter < NN / 256; ++iter) {
    int bcol = iter * 256 + wave * 16 + (lane & 15);
    v8f acc;
#pragma unroll
    for (int g = 0; g < 8; ++g) acc[g] = 0.f;
#pragma unroll
    for (int kc = 0; kc < 4; ++kc) {
      int kb = kc * 32;
      v16bf af = load_frag_a(base + (size_t)arow * CC + kb, lane);
      v16bf bf = load_frag_b(base + (size_t)bcol * CC + kb, lane);
      acc = wmma_bf16(af, bf, acc);
    }
    float sqc = sq[b * NN + bcol];
#pragma unroll
    for (int g = 0; g < 8; ++g)
      dist[g + rhalf][wave * 16 + (lane & 15)] = 2.0f * acc[g] - sqrow[g + rhalf] - sqc;
    __syncthreads();

    // wave w merges the 256 candidates of row w; lane handles cols lane+32*j
#pragma unroll
    for (int j = 0; j < 8; ++j) {
      int cloc = lane + 32 * j;
      float v = dist[wave][cloc];
      int ci = iter * 256 + cloc;
      if (v > minV) {
#pragma unroll
        for (int t = 0; t < KK; ++t)
          if (t == minP) { tval[t] = v; tidx[t] = ci; }
        minV = tval[0]; minP = 0;
#pragma unroll
        for (int t = 1; t < KK; ++t)
          if (tval[t] < minV) { minV = tval[t]; minP = t; }
      }
    }
    __syncthreads();
  }

  // cross-lane extraction of global top-32 (order irrelevant: softmax/sum are k-permutation-invariant)
  int myWin = -1;
  for (int r = 0; r < KK; ++r) {
    float bv = tval[0]; int bp = 0;
#pragma unroll
    for (int t = 1; t < KK; ++t)
      if (tval[t] > bv) { bv = tval[t]; bp = t; }
    int bi = -1;
#pragma unroll
    for (int t = 0; t < KK; ++t)
      if (t == bp) bi = tidx[t];
    float v = bv; int ix = bi;
#pragma unroll
    for (int off = 16; off > 0; off >>= 1) {
      float ov = __shfl_xor(v, off, 32);
      int oi = __shfl_xor(ix, off, 32);
      if (ov > v) { v = ov; ix = oi; }
    }
#pragma unroll
    for (int t = 0; t < KK; ++t)
      if (tidx[t] == ix) { tval[t] = -3.4e38f; tidx[t] = -2; }
    if (lane == r) myWin = ix;
  }
  knn_idx[((size_t)b * NN + rowBase + wave) * KK + lane] = myWin;
}

// ---------------- kernel C: U = pcd @ Wqk^T (bf16 WMMA) ----------------
__global__ void __launch_bounds__(256) ugemm_kernel(const __bf16* __restrict__ pcdh,
                                                    const __bf16* __restrict__ wqk_bf,
                                                    float* __restrict__ U) {
  int b = blockIdx.x / (NN / 16);
  int rowBase = (blockIdx.x % (NN / 16)) * 16;
  int wave = threadIdx.x >> 5;      // 8 waves -> 8 column tiles of 16
  int lane = threadIdx.x & 31;
  int colBase = wave * 16;
  int arow = rowBase + (lane & 15);
  int bcol = colBase + (lane & 15);
  int rhalf = (lane >> 4) << 3;
  const __bf16* abase = pcdh + (size_t)b * NN * CC;

  v8f acc;
#pragma unroll
  for (int g = 0; g < 8; ++g) acc[g] = 0.f;
#pragma unroll
  for (int kc = 0; kc < 4; ++kc) {
    int kb = kc * 32;
    v16bf af = load_frag_a(abase + (size_t)arow * CC + kb, lane);
    v16bf bf = load_frag_b(wqk_bf + (size_t)bcol * CC + kb, lane);
    acc = wmma_bf16(af, bf, acc);
  }
#pragma unroll
  for (int g = 0; g < 8; ++g)
    U[((size_t)b * NN + rowBase + g + rhalf) * CC + colBase + (lane & 15)] = acc[g];
}

// ---------------- kernel D: energies -> softmax -> std score + wdiff ----------------
// one wave per point; lane owns 4 channels; energies live one per lane.
__global__ void __launch_bounds__(256) attn_kernel(const float* __restrict__ pcdf,
                                                   const float* __restrict__ U,
                                                   const int* __restrict__ knn_idx,
                                                   float* __restrict__ sel,
                                                   __bf16* __restrict__ wdiff_bf) {
  int wave = threadIdx.x >> 5;
  int lane = threadIdx.x & 31;
  int pt = blockIdx.x * 8 + wave;          // over B*N
  int b = pt / NN, n = pt % NN;

  const float* ub = U + ((size_t)b * NN + n) * CC + lane * 4;
  const float* pb = pcdf + ((size_t)b * NN + n) * CC + lane * 4;
  v4f u = *(const v4f*)ub;
  v4f p = *(const v4f*)pb;
  const int* nidx = knn_idx + ((size_t)b * NN + n) * KK;
  const float* pcb = pcdf + (size_t)b * NN * CC;

  float myE = 0.f;
#pragma unroll 4
  for (int k = 0; k < KK; ++k) {
    int j = nidx[k];
    v4f nb = *(const v4f*)(pcb + (size_t)j * CC + lane * 4);
    float part = u[0] * (nb[0] - p[0]) + u[1] * (nb[1] - p[1]) +
                 u[2] * (nb[2] - p[2]) + u[3] * (nb[3] - p[3]);
#pragma unroll
    for (int off = 16; off > 0; off >>= 1) part += __shfl_xor(part, off, 32);
    if (lane == k) myE = part;
  }
  // softmax over 32 energies (one per lane)
  float m = myE;
#pragma unroll
  for (int off = 16; off > 0; off >>= 1) m = fmaxf(m, __shfl_xor(m, off, 32));
  float a = __expf(myE - m);
  float s = a;
#pragma unroll
  for (int off = 16; off > 0; off >>= 1) s += __shfl_xor(s, off, 32);
  a /= s;
  float q = a * a;
#pragma unroll
  for (int off = 16; off > 0; off >>= 1) q += __shfl_xor(q, off, 32);
  if (lane == 0) {
    float mean = 1.0f / KK;
    sel[(size_t)b * NN + n] = sqrtf(fmaxf(q / KK - mean * mean, 0.f));
  }
  // wdiff = sum_k a_k * (neighbor - p)
  float w0 = 0.f, w1 = 0.f, w2 = 0.f, w3 = 0.f;
#pragma unroll 4
  for (int k = 0; k < KK; ++k) {
    float ak = __shfl(a, k, 32);
    int j = nidx[k];
    v4f nb = *(const v4f*)(pcb + (size_t)j * CC + lane * 4);
    w0 += ak * (nb[0] - p[0]); w1 += ak * (nb[1] - p[1]);
    w2 += ak * (nb[2] - p[2]); w3 += ak * (nb[3] - p[3]);
  }
  __bf16* wd = wdiff_bf + ((size_t)b * NN + n) * CC + lane * 4;
  wd[0] = (__bf16)w0; wd[1] = (__bf16)w1; wd[2] = (__bf16)w2; wd[3] = (__bf16)w3;
}

// ---------------- kernel E: exact top-M by rank counting (preserves top_k order) ----
__global__ void select_kernel(const float* __restrict__ sel, int* __restrict__ sel_idx) {
  int t = blockIdx.x * blockDim.x + threadIdx.x;  // over B*N
  int b = t / NN, n = t % NN;
  const float* s = sel + (size_t)b * NN;
  float v = s[n];
  int rank = 0;
  for (int m = 0; m < NN; ++m) {
    float w = s[m];
    rank += (w > v) || (w == v && m < n);
  }
  if (rank < MM) sel_idx[b * MM + rank] = n;
}

// ---------------- kernel F: out = Wv @ gather(wdiff)^T (bf16 WMMA) ----------------
__global__ void __launch_bounds__(256) out_kernel(const __bf16* __restrict__ wv_bf,
                                                  const __bf16* __restrict__ wdiff_bf,
                                                  const int* __restrict__ sel_idx,
                                                  float* __restrict__ out) {
  int b = blockIdx.x / (MM / 16);
  int mBase = (blockIdx.x % (MM / 16)) * 16;
  int wave = threadIdx.x >> 5;   // 8 waves -> 8 d-tiles of 16
  int lane = threadIdx.x & 31;
  int dBase = wave * 16;
  int rhalf = (lane >> 4) << 3;
  int arow = dBase + (lane & 15);                       // Wv row (output channel)
  int srow = sel_idx[b * MM + mBase + (lane & 15)];     // gathered point
  const __bf16* bptr = wdiff_bf + ((size_t)b * NN + srow) * CC;

  v8f acc;
#pragma unroll
  for (int g = 0; g < 8; ++g) acc[g] = 0.f;
#pragma unroll
  for (int kc = 0; kc < 4; ++kc) {
    int kb = kc * 32;
    v16bf af = load_frag_a(wv_bf + (size_t)arow * CC + kb, lane);
    v16bf bf = load_frag_b(bptr + kb, lane);
    acc = wmma_bf16(af, bf, acc);
  }
#pragma unroll
  for (int g = 0; g < 8; ++g)
    out[((size_t)b * CC + dBase + g + rhalf) * MM + mBase + (lane & 15)] = acc[g];
}

// ---------------- host launch ----------------
extern "C" void kernel_launch(void* const* d_in, const int* in_sizes, int n_in,
                              void* d_out, int out_size, void* d_ws, size_t ws_size,
                              hipStream_t stream) {
  const float* x  = (const float*)d_in[0];
  const float* Wq = (const float*)d_in[1];
  const float* Wk = (const float*)d_in[2];
  const float* Wv = (const float*)d_in[3];
  float* out = (float*)d_out;

  char* ws = (char*)d_ws;
  size_t off = 0;
  float*  pcdf    = (float*)(ws + off);  off += (size_t)BB * NN * CC * 4;   // 8 MB
  __bf16* pcdh    = (__bf16*)(ws + off); off += (size_t)BB * NN * CC * 2;   // 4 MB
  float*  sq      = (float*)(ws + off);  off += (size_t)BB * NN * 4;        // 64 KB
  __bf16* wqk_bf  = (__bf16*)(ws + off); off += (size_t)CC * CC * 2;        // 32 KB
  __bf16* wv_bf   = (__bf16*)(ws + off); off += (size_t)CC * CC * 2;        // 32 KB
  int*    knn_idx = (int*)(ws + off);    off += (size_t)BB * NN * KK * 4;   // 2 MB
  float*  U       = (float*)(ws + off);  off += (size_t)BB * NN * CC * 4;   // 8 MB
  float*  sel     = (float*)(ws + off);  off += (size_t)BB * NN * 4;        // 64 KB
  __bf16* wdiff   = (__bf16*)(ws + off); off += (size_t)BB * NN * CC * 2;   // 4 MB
  int*    sel_idx = (int*)(ws + off);    off += (size_t)BB * MM * 4;        // 32 KB

  prep_kernel<<<(BB * CC * NN) / 256, 256, 0, stream>>>(x, pcdf, pcdh);
  norm_kernel<<<(BB * NN) / 256, 256, 0, stream>>>(pcdf, sq);
  prep_w_kernel<<<(CC * CC) / 256, 256, 0, stream>>>(Wq, Wk, Wv, wqk_bf, wv_bf);
  knn_kernel<<<BB * (NN / 16), 512, 0, stream>>>(pcdh, sq, knn_idx);
  ugemm_kernel<<<BB * (NN / 16), 256, 0, stream>>>(pcdh, wqk_bf, U);
  attn_kernel<<<(BB * NN) / 8, 256, 0, stream>>>(pcdf, U, knn_idx, sel, wdiff);
  select_kernel<<<(BB * NN) / 256, 256, 0, stream>>>(sel, sel_idx);
  out_kernel<<<BB * (MM / 16), 256, 0, stream>>>(wv_bf, wdiff, sel_idx, out);
}
